// ActorGNN_14181982012208
// MI455X (gfx1250) — compile-verified, matrix operand order checked by reading, not run on hardware
//
#include <hip/hip_runtime.h>

// ---------------------------------------------------------------------------
// 6-layer GAT (GATConv, heads=2, concat) for MI455X / gfx1250.
//  - Dense projections: V_WMMA_F32_16X16X32_BF16. Weights staged in LDS once
//    per block; each wave owns one 16-row M tile, keeps all four A k-step
//    fragments in registers, and sweeps every 16-col N tile from LDS.
//  - Edge softmax/aggregation: atomic-based (float segment-max via
//    signed-max/unsigned-min bit trick, f32 atomic adds).
// ---------------------------------------------------------------------------

#define N_NODES 50000
#define N_EDGES 800000
#define K_DIM   128          // every layer's input dim is 128
#define HEADS   2
#define N_LAYERS 6
#define E_TOT   (N_EDGES + N_NODES)   // with self-loops

typedef __attribute__((ext_vector_type(16))) __bf16        v16bf;
typedef __attribute__((ext_vector_type(8)))  float         v8f;
typedef __attribute__((ext_vector_type(8)))  unsigned int  v8u;

__device__ __forceinline__ unsigned short f2bf(float f) {
    unsigned u = __float_as_uint(f);
    unsigned r = u + 0x7FFFu + ((u >> 16) & 1u);   // round-to-nearest-even
    return (unsigned short)(r >> 16);
}

// Pack two 16-byte chunks (K = +0..7 and K = +16..23 for this lane-half)
// into a 16x32 bf16 WMMA operand fragment.
__device__ __forceinline__ v16bf pack_frag(uint4 lo, uint4 hi) {
    v8u r;
    r[0] = lo.x; r[1] = lo.y; r[2] = lo.z; r[3] = lo.w;
    r[4] = hi.x; r[5] = hi.y; r[6] = hi.z; r[7] = hi.w;
    return __builtin_bit_cast(v16bf, r);
}

// ---------------------------------------------------------------------------
// Conversions
// ---------------------------------------------------------------------------
__global__ void k_f32_to_bf16(const float* __restrict__ in,
                              unsigned short* __restrict__ out, int n) {
    int t = blockIdx.x * blockDim.x + threadIdx.x;
    if (t < n) out[t] = f2bf(in[t]);
}

// W: [K_DIM, nout] f32 row-major  ->  Wt: [nout, K_DIM] bf16 (N-major)
__global__ void k_w_transpose_bf16(const float* __restrict__ W,
                                   unsigned short* __restrict__ Wt, int nout) {
    int t = blockIdx.x * blockDim.x + threadIdx.x;
    if (t >= K_DIM * nout) return;
    int k = t / nout, n2 = t - k * nout;
    Wt[n2 * K_DIM + k] = f2bf(W[t]);
}

// ---------------------------------------------------------------------------
// WMMA GEMM: H[M, NOUT] = Xb[M, K] * Wt[NOUT, K]^T  (bf16 in, f32 out)
// Per-lane fragment layout (16-bit A/B): lanes 0-15 carry K = k0+0..7 and
// k0+16..23, lanes 16-31 carry K = k0+8..15 and k0+24..31.
// Block: 256 threads = 8 waves; Wt staged into LDS once per block.
// ---------------------------------------------------------------------------
template <int NOUT>
__global__ __launch_bounds__(256)
void k_gemm_wmma_bf16(const unsigned short* __restrict__ Xb,
                      const unsigned short* __restrict__ Wt,
                      float* __restrict__ H) {
    __shared__ unsigned short Bs[NOUT * K_DIM];   // 32KB (128) / 64KB (256)

    const int tid = threadIdx.x;

    // Cooperative stage of the bf16 weight matrix into LDS (b128 chunks).
    {
        const uint4* gsrc = reinterpret_cast<const uint4*>(Wt);
        uint4*       ldst = reinterpret_cast<uint4*>(Bs);
        const int nchunks = NOUT * K_DIM / 8;     // 8 shorts per uint4
#pragma unroll
        for (int i = 0; i < nchunks / 256; ++i)
            ldst[i * 256 + tid] = gsrc[i * 256 + tid];
    }
    __syncthreads();

    const int lane    = tid & 31;
    const int wavein  = tid >> 5;
    const int mt      = blockIdx.x * 8 + wavein;  // one 16-row M tile per wave
    if (mt * 16 >= N_NODES) return;

    const int halfsel = lane >> 4;                // K-half select
    const int idx     = lane & 15;                // row (A) / col (B) in tile

    // A fragments for all four k-steps, held in registers.
    const uint4* aq = reinterpret_cast<const uint4*>(
        Xb + (size_t)(mt * 16 + idx) * K_DIM + halfsel * 8);
    v16bf a0 = pack_frag(aq[0],  aq[2]);          // K   0..31
    v16bf a1 = pack_frag(aq[4],  aq[6]);          // K  32..63
    v16bf a2 = pack_frag(aq[8],  aq[10]);         // K  64..95
    v16bf a3 = pack_frag(aq[12], aq[14]);         // K  96..127

#pragma unroll
    for (int nt = 0; nt < NOUT / 16; ++nt) {
        const uint4* bq = reinterpret_cast<const uint4*>(
            Bs + (nt * 16 + idx) * K_DIM + halfsel * 8);
        v8f c = {};
        c = __builtin_amdgcn_wmma_f32_16x16x32_bf16(false, a0, false,
                pack_frag(bq[0],  bq[2]),  (short)0, c, false, false);
        c = __builtin_amdgcn_wmma_f32_16x16x32_bf16(false, a1, false,
                pack_frag(bq[4],  bq[6]),  (short)0, c, false, false);
        c = __builtin_amdgcn_wmma_f32_16x16x32_bf16(false, a2, false,
                pack_frag(bq[8],  bq[10]), (short)0, c, false, false);
        c = __builtin_amdgcn_wmma_f32_16x16x32_bf16(false, a3, false,
                pack_frag(bq[12], bq[14]), (short)0, c, false, false);

        // C/D layout: lane idx = column; VGPR r holds M = r + halfsel*8
        float* crow = H + (size_t)(mt * 16 + halfsel * 8) * NOUT + nt * 16 + idx;
#pragma unroll
        for (int r = 0; r < 8; ++r) crow[(size_t)r * NOUT] = c[r];
    }
}

// ---------------------------------------------------------------------------
// Per-node attention logits + init of emax/denom
// ---------------------------------------------------------------------------
__global__ void k_node_prep(const float* __restrict__ H,
                            const float* __restrict__ att_src,
                            const float* __restrict__ att_dst,
                            float* __restrict__ a_s, float* __restrict__ a_d,
                            float* __restrict__ emax, float* __restrict__ denom,
                            int D) {
    int t = blockIdx.x * blockDim.x + threadIdx.x;
    if (t >= N_NODES * HEADS) return;
    int n = t >> 1, hd = t & 1;
    const float* hrow = H + (size_t)n * (HEADS * D) + hd * D;
    const float* as   = att_src + hd * D;
    const float* ad   = att_dst + hd * D;
    float s1 = 0.f, s2 = 0.f;
    for (int d = 0; d < D; ++d) { float hv = hrow[d]; s1 += hv * as[d]; s2 += hv * ad[d]; }
    a_s[t] = s1;
    a_d[t] = s2;
    ((unsigned int*)emax)[t] = 0xFF800000u;   // -inf bit pattern
    denom[t] = 0.f;
}

__global__ void k_zero(float* __restrict__ p, int n) {
    int t = blockIdx.x * blockDim.x + threadIdx.x;
    if (t < n) p[t] = 0.f;
}

// ---------------------------------------------------------------------------
// Edge passes
// ---------------------------------------------------------------------------
__device__ __forceinline__ void edge_nodes(const int* __restrict__ src,
                                           const int* __restrict__ dst,
                                           int e, int& s, int& d) {
    if (e < N_EDGES) { s = src[e]; d = dst[e]; }
    else             { s = d = e - N_EDGES; }     // self-loop
}

__device__ __forceinline__ float leaky(float v) { return v > 0.f ? v : 0.2f * v; }

__global__ void k_edge_max(const int* __restrict__ src, const int* __restrict__ dst,
                           const float* __restrict__ a_s, const float* __restrict__ a_d,
                           float* __restrict__ emax) {
    int t = blockIdx.x * blockDim.x + threadIdx.x;
    if (t >= E_TOT * HEADS) return;
    int e = t >> 1, hd = t & 1;
    int s, d; edge_nodes(src, dst, e, s, d);
    float v = leaky(a_s[s * HEADS + hd] + a_d[d * HEADS + hd]);
    float* addr = emax + d * HEADS + hd;
    // Monotonic-bits float max: non-negative via signed max, negative via unsigned min.
    if (v >= 0.f) atomicMax((int*)addr,          __float_as_int(v));
    else          atomicMin((unsigned int*)addr, __float_as_uint(v));
}

__global__ void k_edge_expsum(const int* __restrict__ src, const int* __restrict__ dst,
                              const float* __restrict__ a_s, const float* __restrict__ a_d,
                              const float* __restrict__ emax, float* __restrict__ denom) {
    int t = blockIdx.x * blockDim.x + threadIdx.x;
    if (t >= E_TOT * HEADS) return;
    int e = t >> 1, hd = t & 1;
    int s, d; edge_nodes(src, dst, e, s, d);
    float v = leaky(a_s[s * HEADS + hd] + a_d[d * HEADS + hd]);
    float w = __expf(v - emax[d * HEADS + hd]);
    atomicAdd(denom + d * HEADS + hd, w);
}

// One wave (32 lanes) per edge: lanes stride the HD feature vector.
__global__ void k_edge_scatter(const int* __restrict__ src, const int* __restrict__ dst,
                               const float* __restrict__ a_s, const float* __restrict__ a_d,
                               const float* __restrict__ emax, const float* __restrict__ denom,
                               const float* __restrict__ H, float* __restrict__ acc, int D) {
    int wave = (blockIdx.x * blockDim.x + threadIdx.x) >> 5;
    int lane = threadIdx.x & 31;
    if (wave >= E_TOT) return;
    int s, d; edge_nodes(src, dst, wave, s, d);
    const int HD = HEADS * D;

    const float* hs = H + (size_t)s * HD;
    __builtin_prefetch(hs, 0, 1);     // global_prefetch of the gathered row

    float v0 = leaky(a_s[s * 2 + 0] + a_d[d * 2 + 0]);
    float v1 = leaky(a_s[s * 2 + 1] + a_d[d * 2 + 1]);
    float al0 = __expf(v0 - emax[d * 2 + 0]) / (denom[d * 2 + 0] + 1e-16f);
    float al1 = __expf(v1 - emax[d * 2 + 1]) / (denom[d * 2 + 1] + 1e-16f);

    float* ao = acc + (size_t)d * HD;
    for (int f = lane; f < HD; f += 32) {
        float a = (f < D) ? al0 : al1;
        atomicAdd(ao + f, hs[f] * a);
    }
}

// ---------------------------------------------------------------------------
// acc + bias, then relu+bf16 (intermediate) or f32 store (final layer)
// ---------------------------------------------------------------------------
__global__ void k_finalize(const float* __restrict__ acc, const float* __restrict__ bias,
                           unsigned short* __restrict__ Xb_next, float* __restrict__ outF,
                           int HD, int is_last) {
    int t = blockIdx.x * blockDim.x + threadIdx.x;
    if (t >= N_NODES * HD) return;
    float y = acc[t] + bias[t & (HD - 1)];
    if (is_last) {
        outF[t] = y;
    } else {
        y = y > 0.f ? y : 0.f;
        Xb_next[t] = f2bf(y);
    }
}

// ---------------------------------------------------------------------------
static inline int cdiv64(long long a, int b) { return (int)((a + b - 1) / b); }

extern "C" void kernel_launch(void* const* d_in, const int* in_sizes, int n_in,
                              void* d_out, int out_size, void* d_ws, size_t ws_size,
                              hipStream_t stream) {
    (void)in_sizes; (void)n_in; (void)out_size; (void)ws_size;

    const float* x   = (const float*)d_in[0];
    const int*   ei  = (const int*)d_in[1];      // [2, N_EDGES] row-major int32
    const int*   src = ei;
    const int*   dst = ei + N_EDGES;

    // Workspace carve-up (256B aligned), ~117 MB total.
    char*  ws  = (char*)d_ws;
    size_t off = 0;
    auto take = [&](size_t bytes) -> void* {
        void* p = ws + off;
        off += (bytes + 255) & ~(size_t)255;
        return p;
    };
    unsigned short* Xb   = (unsigned short*)take((size_t)N_NODES * K_DIM * 2);
    unsigned short* Wt   = (unsigned short*)take((size_t)256 * K_DIM * 2);
    float* Hbuf  = (float*)take((size_t)N_NODES * 256 * 4);
    float* acc   = (float*)take((size_t)N_NODES * 256 * 4);
    float* a_s   = (float*)take((size_t)N_NODES * HEADS * 4);
    float* a_d   = (float*)take((size_t)N_NODES * HEADS * 4);
    float* emax  = (float*)take((size_t)N_NODES * HEADS * 4);
    float* denom = (float*)take((size_t)N_NODES * HEADS * 4);

    const int B = 256;
    const int mblocks = cdiv64(N_NODES / 16, 8);   // 8 M-tile waves per block

    // Layer-0 input -> bf16
    k_f32_to_bf16<<<cdiv64((long long)N_NODES * K_DIM, B), B, 0, stream>>>(
        x, Xb, N_NODES * K_DIM);

    for (int L = 0; L < N_LAYERS; ++L) {
        const float* W    = (const float*)d_in[2 + 4 * L + 0];
        const float* Asrc = (const float*)d_in[2 + 4 * L + 1];
        const float* Adst = (const float*)d_in[2 + 4 * L + 2];
        const float* Bias = (const float*)d_in[2 + 4 * L + 3];
        const int D  = (L == N_LAYERS - 1) ? 128 : 64;
        const int HD = HEADS * D;

        // W -> bf16, transposed to N-major
        k_w_transpose_bf16<<<cdiv64((long long)K_DIM * HD, B), B, 0, stream>>>(W, Wt, HD);

        // h = x @ W   (WMMA; LDS-staged weights, fully unrolled N sweep)
        if (HD == 128)
            k_gemm_wmma_bf16<128><<<mblocks, B, 0, stream>>>(Xb, Wt, Hbuf);
        else
            k_gemm_wmma_bf16<256><<<mblocks, B, 0, stream>>>(Xb, Wt, Hbuf);

        // per-node logits + init emax/denom; zero accumulator
        k_node_prep<<<cdiv64((long long)N_NODES * HEADS, B), B, 0, stream>>>(
            Hbuf, Asrc, Adst, a_s, a_d, emax, denom, D);
        k_zero<<<cdiv64((long long)N_NODES * HD, B), B, 0, stream>>>(acc, N_NODES * HD);

        // segment softmax + aggregation over edges
        k_edge_max<<<cdiv64((long long)E_TOT * HEADS, B), B, 0, stream>>>(
            src, dst, a_s, a_d, emax);
        k_edge_expsum<<<cdiv64((long long)E_TOT * HEADS, B), B, 0, stream>>>(
            src, dst, a_s, a_d, emax, denom);
        k_edge_scatter<<<cdiv64((long long)E_TOT * 32, B), B, 0, stream>>>(
            src, dst, a_s, a_d, emax, denom, Hbuf, acc, D);

        // bias (+relu+bf16 for intermediate layers; f32 out for last)
        k_finalize<<<cdiv64((long long)N_NODES * HD, B), B, 0, stream>>>(
            acc, Bias, Xb, (float*)d_out, HD, (L == N_LAYERS - 1) ? 1 : 0);
    }
}
